// TransformerBlockCosine_35802847379565
// MI455X (gfx1250) — matrix operand order repaired and checked by reading, not compiled, not run on hardware
//
#include <hip/hip_runtime.h>
#include <hip/hip_bf16.h>
#include <math.h>

#define B_    4
#define N_    4096
#define K_    16
#define DP_   64
#define DM_   128
#define LDP_  132   // f32 LDS row pitch (528B, 16B-aligned)
#define LDH_  136   // f16 LDS row pitch (272B, 16B-aligned)
#define EPS_  1e-8f

typedef __attribute__((ext_vector_type(16))) _Float16 v16h;
typedef __attribute__((ext_vector_type(8)))  _Float16 v8h;
typedef __attribute__((ext_vector_type(4)))  _Float16 v4h;
typedef __attribute__((ext_vector_type(8)))  float    v8f;
typedef __attribute__((ext_vector_type(4)))  float    v4f;

// ---------------- WMMA helpers (gfx1250 wave32, CDNA5 ISA §7.12.2) ----------------

__device__ __forceinline__ v8f wmma_f16(v16h a, v16h b, v8f c) {
  // D = A(16x32 f16) * B(32x16 f16) + C(16x16 f32)
  return __builtin_amdgcn_wmma_f32_16x16x32_f16(false, a, false, b, (short)0, c, false, false);
}

// A-fragment from f16 row-major memory (global): lane r=lane&15, K halves
// [kb..kb+7] and [16+kb..16+kb+7] -> two 16B contiguous loads.
__device__ __forceinline__ v16h afrag_f16g(const _Float16* __restrict__ A, int lda, int lane) {
  const int r = lane & 15, kb = (lane >> 4) * 8;
  const _Float16* p = A + (size_t)r * lda + kb;
  v8h lo = *(const v8h*)__builtin_assume_aligned(p, 16);
  v8h hi = *(const v8h*)__builtin_assume_aligned(p + 16, 16);
  return __builtin_shufflevector(lo, hi, 0,1,2,3,4,5,6,7,8,9,10,11,12,13,14,15);
}

// A-fragment from f16 LDS staging buffer (pitch LDH_), k-step s.
__device__ __forceinline__ v16h afrag_lds(const _Float16* th, int s, int lane) {
  const int r = lane & 15, kb = (lane >> 4) * 8;
  const _Float16* p = th + r * LDH_ + s * 32 + kb;
  v8h lo = *(const v8h*)__builtin_assume_aligned(p, 16);
  v8h hi = *(const v8h*)__builtin_assume_aligned(p + 16, 16);
  return __builtin_shufflevector(lo, hi, 0,1,2,3,4,5,6,7,8,9,10,11,12,13,14,15);
}

// A-fragment from f32 global (converted on load).
__device__ __forceinline__ v16h afrag_f32g(const float* __restrict__ A, int lda, int lane) {
  const int r = lane & 15, kb = (lane >> 4) * 8;
  const float* p = A + (size_t)r * lda + kb;
  v4f f0 = *(const v4f*)__builtin_assume_aligned(p, 16);
  v4f f1 = *(const v4f*)__builtin_assume_aligned(p + 4, 16);
  v4f f2 = *(const v4f*)__builtin_assume_aligned(p + 16, 16);
  v4f f3 = *(const v4f*)__builtin_assume_aligned(p + 20, 16);
  v16h a;
#pragma unroll
  for (int j = 0; j < 4; ++j) {
    a[j]      = (_Float16)f0[j];
    a[4 + j]  = (_Float16)f1[j];
    a[8 + j]  = (_Float16)f2[j];
    a[12 + j] = (_Float16)f3[j];
  }
  return a;
}

// B-fragment (32x16): weights stored TRANSPOSED (N x K row-major), so each lane's
// K-values for its column are row-contiguous -> two 16B loads.
// col = coloff + (lane&15), K rows sbase+kb..+7 and sbase+16+kb..+7 (kb = 0 / 8).
__device__ __forceinline__ v16h bfrag_t(const _Float16* __restrict__ Wt, int Kd,
                                        int coloff, int sbase, int lane) {
  const int cl = lane & 15, kb = (lane >> 4) * 8;
  const _Float16* p = Wt + (size_t)(coloff + cl) * Kd + sbase + kb;
  v8h lo = *(const v8h*)__builtin_assume_aligned(p, 16);
  v8h hi = *(const v8h*)__builtin_assume_aligned(p + 16, 16);
  return __builtin_shufflevector(lo, hi, 0,1,2,3,4,5,6,7,8,9,10,11,12,13,14,15);
}

// ---------------- weight conversion: f32 KxN -> f16 NxK (transposed) ----------------

__global__ void cvt_t_kernel(const float* __restrict__ src, _Float16* __restrict__ dst,
                             int Krows, int Ncols) {
  int i = blockIdx.x * 256 + threadIdx.x;
  if (i < Krows * Ncols) {
    int k = i / Ncols, n = i - k * Ncols;
    dst[(size_t)n * Krows + k] = (_Float16)src[i];
  }
}

// ---------------- KNN: top-16 via register insertion sort ----------------

__global__ void __launch_bounds__(256) knn_kernel(const float* __restrict__ xyz,
                                                  int* __restrict__ knn) {
  __shared__ float sx[256], sy[256], sz[256], ss[256];
  const int b = blockIdx.x >> 4;
  const int n = ((blockIdx.x & 15) << 8) + threadIdx.x;
  const float px = xyz[((size_t)b * N_ + n) * 3 + 0];
  const float py = xyz[((size_t)b * N_ + n) * 3 + 1];
  const float pz = xyz[((size_t)b * N_ + n) * 3 + 2];
  const float ps = px * px + py * py + pz * pz;
  float bd[K_]; int bi[K_];
#pragma unroll
  for (int i = 0; i < K_; ++i) { bd[i] = 3.4e38f; bi[i] = 0; }
  for (int jt = 0; jt < N_; jt += 256) {
    __syncthreads();
    {
      const float* q = xyz + ((size_t)b * N_ + jt + threadIdx.x) * 3;
      float qx = q[0], qy = q[1], qz = q[2];
      sx[threadIdx.x] = qx; sy[threadIdx.x] = qy; sz[threadIdx.x] = qz;
      ss[threadIdx.x] = qx * qx + qy * qy + qz * qz;
    }
    __syncthreads();
    for (int j = 0; j < 256; ++j) {
      // same formula as reference: |p|^2 + |q|^2 - 2 p.q
      float d = ps + ss[j] - 2.f * (px * sx[j] + py * sy[j] + pz * sz[j]);
      if (d < bd[K_ - 1]) {
        bd[K_ - 1] = d; bi[K_ - 1] = jt + j;
#pragma unroll
        for (int s = K_ - 1; s > 0; --s)
          if (bd[s] < bd[s - 1]) {
            float td = bd[s]; bd[s] = bd[s - 1]; bd[s - 1] = td;
            int   ti = bi[s]; bi[s] = bi[s - 1]; bi[s - 1] = ti;
          }
      }
    }
  }
#pragma unroll
  for (int i = 0; i < K_; ++i) knn[((size_t)b * N_ + n) * K_ + i] = bi[i];
}

// ---------------- fc1: x = features @ fc1_w + b  (16384x64 @ 64x128, f16 out) ----------------

__global__ void __launch_bounds__(128) fc1_kernel(const float* __restrict__ feat,
                                                  const _Float16* __restrict__ w_t,  // 128x64
                                                  const float* __restrict__ bias,
                                                  _Float16* __restrict__ x_h) {
  const int lane = threadIdx.x & 31;
  const int rt = blockIdx.x * 4 + (threadIdx.x >> 5);  // 16-row tile, 0..1023
  const float* A = feat + (size_t)rt * 16 * DP_;
  const v16h a0 = afrag_f32g(A, DP_, lane);
  const v16h a1 = afrag_f32g(A + 32, DP_, lane);
  const int cl = lane & 15, mb = (lane >> 4) * 8;
#pragma unroll
  for (int t = 0; t < 8; ++t) {
    v8f c = {};
    c = wmma_f16(a0, bfrag_t(w_t, DP_, t * 16, 0, lane), c);
    c = wmma_f16(a1, bfrag_t(w_t, DP_, t * 16, 32, lane), c);
    const int col = t * 16 + cl;
    const float bb = bias[col];
#pragma unroll
    for (int v = 0; v < 8; ++v)
      x_h[((size_t)rt * 16 + mb + v) * DM_ + col] = (_Float16)(c[v] + bb);
  }
}

// ---------------- q/k/v: 3x (16384x128 @ 128x128, f32 out) ----------------

__device__ __forceinline__ void proj_store(const v16h a[4], const _Float16* __restrict__ Wt,
                                           float* __restrict__ O, size_t row0, int lane) {
  const int cl = lane & 15, mb = (lane >> 4) * 8;
#pragma unroll
  for (int t = 0; t < 8; ++t) {
    v8f c = {};
#pragma unroll
    for (int s = 0; s < 4; ++s)
      c = wmma_f16(a[s], bfrag_t(Wt, DM_, t * 16, s * 32, lane), c);
    const int col = t * 16 + cl;
#pragma unroll
    for (int v = 0; v < 8; ++v) O[(row0 + mb + v) * DM_ + col] = c[v];
  }
}

__global__ void __launch_bounds__(128) qkv_kernel(const _Float16* __restrict__ x_h,
                                                  const _Float16* __restrict__ wq_t,
                                                  const _Float16* __restrict__ wk_t,
                                                  const _Float16* __restrict__ wv_t,
                                                  float* __restrict__ q_f,
                                                  float* __restrict__ k_f,
                                                  float* __restrict__ v_f) {
  const int lane = threadIdx.x & 31;
  const int rt = blockIdx.x * 4 + (threadIdx.x >> 5);
  const _Float16* A = x_h + (size_t)rt * 16 * DM_;
  v16h a[4];
#pragma unroll
  for (int s = 0; s < 4; ++s) a[s] = afrag_f16g(A + s * 32, DM_, lane);
  __builtin_prefetch(wk_t, 0, 2);   // global_prefetch_b8: next weight stream
  proj_store(a, wq_t, q_f, (size_t)rt * 16, lane);
  proj_store(a, wk_t, k_f, (size_t)rt * 16, lane);
  proj_store(a, wv_t, v_f, (size_t)rt * 16, lane);
}

// ---------------- fused per-point attention (wave-per-point) ----------------

struct alignas(32) WaveScratch {
  float    kf[16 * LDP_];   // gathered k rows (f32)
  float    pe[16 * LDP_];   // pos_enc result (f32)
  float    tb[16 * LDP_];   // f32 working buffer (softmax input / output)
  _Float16 th[16 * LDH_];   // f16 GEMM-A staging
  float    qrow[DM_];
  float    resrow[DM_];
  float    kn[16];
  float    sim[16];
  float    rel[16 * 3];
  float    xyzq[3];
  float    qn;
  int      idx[16];
};

// acc[t] = A(16x128 f16 in LDS th) @ Wt(transposed 128x128 f16 in global)
__device__ __forceinline__ void gemm_lds(const _Float16* th,
                                         const _Float16* __restrict__ Wt,
                                         v8f acc[8], int lane) {
  v16h a[4];
#pragma unroll
  for (int s = 0; s < 4; ++s) a[s] = afrag_lds(th, s, lane);
  __builtin_prefetch(Wt, 0, 2);
#pragma unroll
  for (int t = 0; t < 8; ++t) {
    v8f c = {};
#pragma unroll
    for (int s = 0; s < 4; ++s)
      c = wmma_f16(a[s], bfrag_t(Wt, DM_, t * 16, s * 32, lane), c);
    acc[t] = c;
  }
}

__global__ void __launch_bounds__(64) attn_kernel(
    const float* __restrict__ xyz, const float* __restrict__ feat,
    const int* __restrict__ knn,
    const float* __restrict__ q_f, const float* __restrict__ k_f, const float* __restrict__ v_f,
    const _Float16* __restrict__ d2_t, const _Float16* __restrict__ simw_t,
    const _Float16* __restrict__ g1_t, const _Float16* __restrict__ g2_t,
    const float* __restrict__ d1_w, const float* __restrict__ d1_b,
    const float* __restrict__ d2_b, const float* __restrict__ sim_w0,
    const float* __restrict__ sim_b, const float* __restrict__ g1_b,
    const float* __restrict__ g2_b, const float* __restrict__ fc2_w,
    const float* __restrict__ fc2_b,
    float* __restrict__ out_res, float* __restrict__ out_attn) {
  __shared__ WaveScratch wsS[2];
  const int wv   = threadIdx.x >> 5;
  const int lane = threadIdx.x & 31;
  WaveScratch& S = wsS[wv];
  const int p = blockIdx.x * 2 + wv;        // flat point id, 0..B*N-1
  const int b = p >> 12;                    // / 4096
  const int cl = lane & 15, mb = (lane >> 4) * 8;
  const int c0 = lane * 4;                  // 4 contiguous channels per lane

  // ---- stage neighbor ids, q row, query xyz ----
  if (lane < 16) S.idx[lane] = knn[(size_t)p * K_ + lane];
  if (lane < 3)  S.xyzq[lane] = xyz[(size_t)p * 3 + lane];
  ((v4f*)S.qrow)[lane] = ((const v4f*)(q_f + (size_t)p * DM_))[lane];
  __syncthreads();

  // ---- gather k rows (vectorized b128) ----
  for (int r = 0; r < K_; ++r) {
    const v4f kv = ((const v4f*)(k_f + ((size_t)b * N_ + S.idx[r]) * DM_))[lane];
    *(v4f*)&S.kf[r * LDP_ + c0] = kv;
  }
  __syncthreads();

  // ---- norms + relative positions ----
  if (lane < 16) {
    float s = 0.f;
    for (int c = 0; c < DM_; ++c) { float kv = S.kf[lane * LDP_ + c]; s += kv * kv; }
    S.kn[lane] = fmaxf(sqrtf(s), EPS_);
    const float* nx = xyz + ((size_t)b * N_ + S.idx[lane]) * 3;
#pragma unroll
    for (int t = 0; t < 3; ++t) S.rel[lane * 3 + t] = S.xyzq[t] - nx[t];
  } else if (lane == 16) {
    float s = 0.f;
    for (int c = 0; c < DM_; ++c) { float qv = S.qrow[c]; s += qv * qv; }
    S.qn = fmaxf(sqrtf(s), EPS_);
  }
  __syncthreads();

  // ---- cosine similarity ----
  if (lane < 16) {
    float d = 0.f;
    for (int c = 0; c < DM_; ++c) d += S.qrow[c] * S.kf[lane * LDP_ + c];
    S.sim[lane] = d / (S.qn * S.kn[lane]);
  }
  __syncthreads();

  // ---- pos-enc layer 1 (K=3, VALU): th = f16(relu(rel @ d1 + d1_b)) ----
  {
    const v4f w0 = *(const v4f*)(d1_w + c0);
    const v4f w1 = *(const v4f*)(d1_w + DM_ + c0);
    const v4f w2 = *(const v4f*)(d1_w + 2 * DM_ + c0);
    const v4f bb = *(const v4f*)(d1_b + c0);
    for (int r = 0; r < K_; ++r) {
      const float rx = S.rel[r * 3 + 0], ry = S.rel[r * 3 + 1], rz = S.rel[r * 3 + 2];
      v4h hv;
#pragma unroll
      for (int j = 0; j < 4; ++j) {
        float h = fmaf(rx, w0[j], fmaf(ry, w1[j], fmaf(rz, w2[j], bb[j])));
        hv[j] = (_Float16)fmaxf(h, 0.f);
      }
      *(v4h*)&S.th[r * LDH_ + c0] = hv;
    }
  }
  __syncthreads();

  v8f acc[8];

  // ---- pe = th @ d2 + d2_b (f32) ----
  gemm_lds(S.th, d2_t, acc, lane);
#pragma unroll
  for (int t = 0; t < 8; ++t) {
    const int col = t * 16 + cl;
    const float bb = d2_b[col];
#pragma unroll
    for (int v = 0; v < 8; ++v) S.pe[(mb + v) * LDP_ + col] = acc[t][v] + bb;
  }
  __syncthreads();

  // ---- th = f16(q - k) ----
  {
    const v4f q4 = *(const v4f*)&S.qrow[c0];
    for (int r = 0; r < K_; ++r) {
      const v4f k4 = *(const v4f*)&S.kf[r * LDP_ + c0];
      v4h dv;
#pragma unroll
      for (int j = 0; j < 4; ++j) dv[j] = (_Float16)(q4[j] - k4[j]);
      *(v4h*)&S.th[r * LDH_ + c0] = dv;
    }
  }
  __syncthreads();

  // ---- th = f16((q-k)@sim_w[1:] + sim*sim_w[0] + sim_b + pe) (129th K rank-1) ----
  gemm_lds(S.th, simw_t, acc, lane);
#pragma unroll
  for (int t = 0; t < 8; ++t) {
    const int col = t * 16 + cl;
    const float w0 = sim_w0[col], bb = sim_b[col];
#pragma unroll
    for (int v = 0; v < 8; ++v) {
      const int m = mb + v;
      S.th[m * LDH_ + col] =
          (_Float16)(acc[t][v] + S.sim[m] * w0 + bb + S.pe[m * LDP_ + col]);
    }
  }
  __syncthreads();

  // ---- th = f16(relu(th @ g1 + g1_b)) ----
  gemm_lds(S.th, g1_t, acc, lane);
#pragma unroll
  for (int t = 0; t < 8; ++t) {
    const int col = t * 16 + cl;
    const float bb = g1_b[col];
#pragma unroll
    for (int v = 0; v < 8; ++v)
      S.th[(mb + v) * LDH_ + col] = (_Float16)fmaxf(acc[t][v] + bb, 0.f);
  }
  __syncthreads();

  // ---- tb = th @ g2 + g2_b (f32, softmax input) ----
  gemm_lds(S.th, g2_t, acc, lane);
#pragma unroll
  for (int t = 0; t < 8; ++t) {
    const int col = t * 16 + cl;
    const float bb = g2_b[col];
#pragma unroll
    for (int v = 0; v < 8; ++v)
      S.tb[(mb + v) * LDP_ + col] = acc[t][v] + bb;
  }
  __syncthreads();

  // ---- softmax over K axis per channel + weighted (v+pe) reduction ----
  const float isr = 0.08838834764831845f;  // 1/sqrt(128)
  {
    v4f mx = { -3.4e38f, -3.4e38f, -3.4e38f, -3.4e38f };
    for (int r = 0; r < K_; ++r) {
      const v4f x = *(const v4f*)&S.tb[r * LDP_ + c0];
#pragma unroll
      for (int j = 0; j < 4; ++j) mx[j] = fmaxf(mx[j], x[j] * isr);
    }
    v4f sum = { 0.f, 0.f, 0.f, 0.f };
    for (int r = 0; r < K_; ++r) {
      v4f x = *(const v4f*)&S.tb[r * LDP_ + c0];
#pragma unroll
      for (int j = 0; j < 4; ++j) { x[j] = __expf(x[j] * isr - mx[j]); sum[j] += x[j]; }
      *(v4f*)&S.tb[r * LDP_ + c0] = x;
    }
    v4f inv;
#pragma unroll
    for (int j = 0; j < 4; ++j) inv[j] = 1.f / sum[j];
    v4f racc = { 0.f, 0.f, 0.f, 0.f };
    for (int r = 0; r < K_; ++r) {
      v4f a = *(const v4f*)&S.tb[r * LDP_ + c0];
      const v4f vv = *(const v4f*)(v_f + ((size_t)b * N_ + S.idx[r]) * DM_ + c0);
      const v4f pv = *(const v4f*)&S.pe[r * LDP_ + c0];
#pragma unroll
      for (int j = 0; j < 4; ++j) {
        a[j] *= inv[j];
        racc[j] = fmaf(a[j], vv[j] + pv[j], racc[j]);
      }
      *(v4f*)(out_attn + ((size_t)p * K_ + r) * DM_ + c0) = a;
    }
    *(v4f*)&S.resrow[c0] = racc;
  }
  __syncthreads();

  // ---- out = res @ fc2_w + fc2_b + features ----
  for (int f = lane; f < DP_; f += 32) {
    float accv = fc2_b[f];
    for (int c = 0; c < DM_; ++c) accv = fmaf(S.resrow[c], fc2_w[(size_t)c * DP_ + f], accv);
    out_res[(size_t)p * DP_ + f] = accv + feat[(size_t)p * DP_ + f];
  }
}

// ---------------- launch ----------------

extern "C" void kernel_launch(void* const* d_in, const int* in_sizes, int n_in,
                              void* d_out, int out_size, void* d_ws, size_t ws_size,
                              hipStream_t stream) {
  (void)in_sizes; (void)n_in; (void)out_size; (void)ws_size;
  const float* xyz   = (const float*)d_in[0];
  const float* feat  = (const float*)d_in[1];
  const float* fc1_w = (const float*)d_in[2];
  const float* fc1_b = (const float*)d_in[3];
  const float* fc2_w = (const float*)d_in[4];
  const float* fc2_b = (const float*)d_in[5];
  const float* d1_w  = (const float*)d_in[6];
  const float* d1_b  = (const float*)d_in[7];
  const float* d2_w  = (const float*)d_in[8];
  const float* d2_b  = (const float*)d_in[9];
  const float* g1_w  = (const float*)d_in[10];
  const float* g1_b  = (const float*)d_in[11];
  const float* g2_w  = (const float*)d_in[12];
  const float* g2_b  = (const float*)d_in[13];
  const float* wq_w  = (const float*)d_in[14];
  const float* wk_w  = (const float*)d_in[15];
  const float* wv_w  = (const float*)d_in[16];
  const float* sim_w = (const float*)d_in[17];
  const float* sim_b = (const float*)d_in[18];

  // workspace layout
  char* ws = (char*)d_ws;
  const size_t OFF_KNN = 0;
  const size_t OFF_XH  = OFF_KNN + (size_t)B_ * N_ * K_ * sizeof(int);
  const size_t OFF_Q   = OFF_XH  + (size_t)B_ * N_ * DM_ * sizeof(_Float16);
  const size_t OFF_K   = OFF_Q   + (size_t)B_ * N_ * DM_ * sizeof(float);
  const size_t OFF_V   = OFF_K   + (size_t)B_ * N_ * DM_ * sizeof(float);
  const size_t OFF_W   = OFF_V   + (size_t)B_ * N_ * DM_ * sizeof(float);
  int*      knn  = (int*)(ws + OFF_KNN);
  _Float16* x_h  = (_Float16*)(ws + OFF_XH);
  float*    q_f  = (float*)(ws + OFF_Q);
  float*    k_f  = (float*)(ws + OFF_K);
  float*    v_f  = (float*)(ws + OFF_V);
  _Float16* wbase  = (_Float16*)(ws + OFF_W);  // all transposed (N x K) f16
  _Float16* fc1_t  = wbase;                    // 128x64
  _Float16* wq_t   = fc1_t  + DP_ * DM_;       // 128x128 each below
  _Float16* wk_t   = wq_t   + DM_ * DM_;
  _Float16* wv_t   = wk_t   + DM_ * DM_;
  _Float16* d2_t   = wv_t   + DM_ * DM_;
  _Float16* g1_t   = d2_t   + DM_ * DM_;
  _Float16* g2_t   = g1_t   + DM_ * DM_;
  _Float16* simw_t = g2_t   + DM_ * DM_;       // transpose of sim_w rows 1..128

  // transpose+convert weights to f16 (stay hot in the 192MB L2)
  cvt_t_kernel<<<(DP_ * DM_ + 255) / 256, 256, 0, stream>>>(fc1_w, fc1_t, DP_, DM_);
  cvt_t_kernel<<<(DM_ * DM_ + 255) / 256, 256, 0, stream>>>(wq_w, wq_t, DM_, DM_);
  cvt_t_kernel<<<(DM_ * DM_ + 255) / 256, 256, 0, stream>>>(wk_w, wk_t, DM_, DM_);
  cvt_t_kernel<<<(DM_ * DM_ + 255) / 256, 256, 0, stream>>>(wv_w, wv_t, DM_, DM_);
  cvt_t_kernel<<<(DM_ * DM_ + 255) / 256, 256, 0, stream>>>(d2_w, d2_t, DM_, DM_);
  cvt_t_kernel<<<(DM_ * DM_ + 255) / 256, 256, 0, stream>>>(g1_w, g1_t, DM_, DM_);
  cvt_t_kernel<<<(DM_ * DM_ + 255) / 256, 256, 0, stream>>>(g2_w, g2_t, DM_, DM_);
  cvt_t_kernel<<<(DM_ * DM_ + 255) / 256, 256, 0, stream>>>(sim_w + DM_, simw_t, DM_, DM_);

  // KNN selection
  knn_kernel<<<B_ * (N_ / 256), 256, 0, stream>>>(xyz, knn);

  // fc1 projection (WMMA)
  fc1_kernel<<<(B_ * N_ / 16) / 4, 128, 0, stream>>>(feat, fc1_t, fc1_b, x_h);

  // q/k/v projections (WMMA)
  qkv_kernel<<<(B_ * N_ / 16) / 4, 128, 0, stream>>>(x_h, wq_t, wk_t, wv_t, q_f, k_f, v_f);

  // fused attention: one wave per point, 2 points per block
  float* out_res  = (float*)d_out;
  float* out_attn = out_res + (size_t)B_ * N_ * DP_;
  attn_kernel<<<B_ * N_ / 2, 64, 0, stream>>>(
      xyz, feat, knn, q_f, k_f, v_f,
      d2_t, simw_t, g1_t, g2_t,
      d1_w, d1_b, d2_b, sim_w /*row 0*/, sim_b, g1_b, g2_b, fc2_w, fc2_b,
      out_res, out_attn);
}
